// LinearAttention_60078002537061
// MI455X (gfx1250) — matrix operand order repaired and checked by reading, not compiled, not run on hardware
//
#include <hip/hip_runtime.h>
#include <hip/hip_bf16.h>

// ---------------------------------------------------------------------------
// CDNA5 (gfx1250) wave32 WMMA implementation of the linear-attention block.
// All matrix contractions use v_wmma_f32_16x16x32_bf16 (bf16 operands, f32 acc).
// LDS tiles use 16B-aligned pitches so fragment reads lower to ds_load_b128.
// ---------------------------------------------------------------------------

typedef __attribute__((ext_vector_type(16))) __bf16 bf16x16;
typedef __attribute__((ext_vector_type(8)))  float  f32x8;

struct U4 { unsigned x, y, z, w; };   // trivially-copyable 16B chunk

union Frag16 {
  bf16x16        v;
  unsigned short u[16];
  U4             q[2];
};

__device__ __forceinline__ unsigned short f2bf(float f) {
  unsigned u = __float_as_uint(f);
  u += 0x7FFFu + ((u >> 16) & 1u);      // round-to-nearest-even
  return (unsigned short)(u >> 16);
}

__device__ __forceinline__ f32x8 wmma_bf16(const Frag16& a, const Frag16& b, f32x8 c) {
  // (neg_a, A, neg_b, B, c_mod, C, reuse_a, reuse_b)
  return __builtin_amdgcn_wmma_f32_16x16x32_bf16(false, a.v, false, b.v,
                                                 (short)0, c, false, false);
}

// ---------------------------------------------------------------------------
// 1) Channel LayerNorm over C=512, output bf16 in [B][512][4096] layout.
// ---------------------------------------------------------------------------
__global__ __launch_bounds__(256) void k_ln(const float* __restrict__ fmap,
                                            const float* __restrict__ gamma,
                                            unsigned short* __restrict__ fm_bf) {
  const int p  = blockIdx.x * 256 + threadIdx.x;   // pixel id 0..32767
  const int b  = p >> 12;
  const int hw = p & 4095;
  const float* src = fmap + (size_t)b * 512 * 4096 + hw;
  float s = 0.f, s2 = 0.f;
  for (int c = 0; c < 512; ++c) {
    float v = src[(size_t)c * 4096];
    s += v; s2 += v * v;
  }
  const float mean = s * (1.f / 512.f);
  const float var  = s2 * (1.f / 512.f) - mean * mean;
  const float inv  = rsqrtf(var + 1e-5f);
  unsigned short* dst = fm_bf + (size_t)b * 512 * 4096 + hw;
  for (int c = 0; c < 512; ++c) {
    float v = (src[(size_t)c * 4096] - mean) * inv * gamma[c];
    dst[(size_t)c * 4096] = f2bf(v);
  }
}

// ---------------------------------------------------------------------------
// 2/8) Channel GEMM (1x1 conv): Y[b][m][n] = sum_c W[m][c] * X[b][c][n]
//      W fp32 (selected among w0/w1/w2 by output-channel range), X bf16.
//      Block tile 128x128, K-step 32, 8 waves x (4x2) WMMA tiles.
// ---------------------------------------------------------------------------
__global__ __launch_bounds__(256) void k_gemm_wmma(const float* __restrict__ w0,
                                                   const float* __restrict__ w1,
                                                   const float* __restrict__ w2,
                                                   const unsigned short* __restrict__ Xbf,
                                                   float* __restrict__ Y, int Mtotal) {
  __shared__ alignas(16) unsigned short As[128][40];   // [m][k], 80B pitch
  __shared__ alignas(16) unsigned short Bs[128][40];   // [n][k], 80B pitch
  const int tid = threadIdx.x;
  const int b   = blockIdx.z;
  const int m0  = blockIdx.y * 128;
  const int n0  = blockIdx.x * 128;
  const unsigned short* X = Xbf + (size_t)b * 512 * 4096;

  const int lane = tid & 31;
  const int wv   = tid >> 5;            // wave 0..7
  const int mw   = (wv & 1) * 64;
  const int nw   = (wv >> 1) * 32;

  f32x8 acc[4][2];
  f32x8 zero = {};
  for (int i = 0; i < 4; ++i)
    for (int j = 0; j < 2; ++j) acc[i][j] = zero;

  // A staging: thread -> (row, 16-wide k half)
  const int arow = tid >> 1;
  const int akh  = (tid & 1) * 16;
  const int gm   = m0 + arow;
  const float* wsel = (gm < 512) ? w0 : ((gm < 1024) ? w1 : w2);
  const float* asrc = wsel + (size_t)(gm & 511) * 512 + akh;
  // B staging: thread -> (k row, 16 contiguous n)
  const int bkk = tid >> 3;             // 0..31
  const int bnq = (tid & 7) * 16;

  for (int k0 = 0; k0 < 512; k0 += 32) {
    #pragma unroll
    for (int j = 0; j < 16; ++j)
      As[arow][akh + j] = f2bf(asrc[k0 + j]);
    {
      Frag16 t;
      const U4* p = (const U4*)(X + (size_t)(k0 + bkk) * 4096 + n0 + bnq);
      t.q[0] = p[0]; t.q[1] = p[1];
      #pragma unroll
      for (int j = 0; j < 16; ++j)
        Bs[bnq + j][bkk] = t.u[j];
    }
    if (k0 + 32 < 512) {   // prefetch next K tile (global_prefetch_b8 path)
      __builtin_prefetch(asrc + k0 + 32, 0, 0);
      __builtin_prefetch(X + (size_t)(k0 + 32 + bkk) * 4096 + n0 + bnq, 0, 0);
    }
    __syncthreads();

    const int kb  = (lane < 16) ? 0 : 16;  // A lane K-base
    const int kbb = (lane < 16) ? 0 : 16;  // B lane K-base
    Frag16 bfr[2];
    #pragma unroll
    for (int nt = 0; nt < 2; ++nt) {
      const int col = nw + nt * 16 + (lane & 15);
      const U4* p = (const U4*)&Bs[col][kbb];
      bfr[nt].q[0] = p[0]; bfr[nt].q[1] = p[1];
    }
    #pragma unroll
    for (int mt = 0; mt < 4; ++mt) {
      Frag16 af;
      const int row = mw + mt * 16 + (lane & 15);
      af.q[0] = *(const U4*)&As[row][kb];
      af.q[1] = *(const U4*)&As[row][8 + kb];
      #pragma unroll
      for (int nt = 0; nt < 2; ++nt)
        acc[mt][nt] = wmma_bf16(af, bfr[nt], acc[mt][nt]);
    }
    __syncthreads();
  }

  // C/D layout: VGPR r -> M = r + (lane>=16 ? 8 : 0), N = lane & 15
  const int mofs = (lane >> 4) << 3;
  const int ncol = lane & 15;
  #pragma unroll
  for (int mt = 0; mt < 4; ++mt)
    #pragma unroll
    for (int nt = 0; nt < 2; ++nt)
      #pragma unroll
      for (int r = 0; r < 8; ++r) {
        int om = m0 + mw + mt * 16 + r + mofs;
        int on = n0 + nw + nt * 16 + ncol;
        Y[((size_t)b * Mtotal + om) * 4096 + on] = acc[mt][nt][r];
      }
}

// ---------------------------------------------------------------------------
// 3) Depthwise 3x3 'SAME' conv on fused q1/k1/v1 [B][1536][64][64].
//    q,k out as fp32 (softmax follows), v straight to bf16.
// ---------------------------------------------------------------------------
__global__ __launch_bounds__(256) void k_dwconv(const float* __restrict__ qkv1,
                                                const float* __restrict__ wq_dw,
                                                const float* __restrict__ wk_dw,
                                                const float* __restrict__ wv_dw,
                                                float* __restrict__ qf,
                                                float* __restrict__ kf,
                                                unsigned short* __restrict__ v_bf) {
  const int bc = blockIdx.x;            // 0 .. 8*1536-1
  const int b  = bc / 1536;
  const int c  = bc % 1536;
  const int hw = blockIdx.y * 256 + threadIdx.x;
  const int xx = hw >> 6, yy = hw & 63;
  const float* wsel = (c < 512) ? wq_dw : ((c < 1024) ? wk_dw : wv_dw);
  const int cc = c & 511;
  const float* w9  = wsel + cc * 9;
  const float* src = qkv1 + ((size_t)b * 1536 + c) * 4096;
  float sum = 0.f;
  #pragma unroll
  for (int dy = 0; dy < 3; ++dy) {
    int xs = xx + dy - 1;
    if ((unsigned)xs >= 64u) continue;
    #pragma unroll
    for (int dx = 0; dx < 3; ++dx) {
      int ys = yy + dx - 1;
      if ((unsigned)ys >= 64u) continue;
      sum += src[xs * 64 + ys] * w9[dy * 3 + dx];
    }
  }
  const size_t o = ((size_t)b * 512 + cc) * 4096 + hw;
  if (c < 512)       qf[o] = sum;
  else if (c < 1024) kf[o] = sum;
  else               v_bf[o] = f2bf(sum);
}

// ---------------------------------------------------------------------------
// 4) q softmax over feature dim d=64 (per b,h,token), * d^-0.5, -> bf16.
// ---------------------------------------------------------------------------
__global__ __launch_bounds__(256) void k_softmax_q(const float* __restrict__ qf,
                                                   unsigned short* __restrict__ q_bf) {
  const int bh = blockIdx.x;                       // 0..63
  const int n  = blockIdx.y * 256 + threadIdx.x;   // token 0..4095
  const int b = bh >> 3, h = bh & 7;
  const float* base = qf + ((size_t)b * 512 + h * 64) * 4096 + n;
  float m = -1e30f;
  for (int d = 0; d < 64; ++d) m = fmaxf(m, base[(size_t)d * 4096]);
  float s = 0.f;
  for (int d = 0; d < 64; ++d) s += __expf(base[(size_t)d * 4096] - m);
  const float inv = 0.125f / s;                    // scale = 64^-0.5
  unsigned short* dst = q_bf + ((size_t)b * 512 + h * 64) * 4096 + n;
  for (int d = 0; d < 64; ++d)
    dst[(size_t)d * 4096] = f2bf(__expf(base[(size_t)d * 4096] - m) * inv);
}

// ---------------------------------------------------------------------------
// 5) k softmax over token dim n=4096 (per b,channel), -> bf16.
// ---------------------------------------------------------------------------
__global__ __launch_bounds__(256) void k_softmax_k(const float* __restrict__ kf,
                                                   unsigned short* __restrict__ k_bf) {
  __shared__ float red[256];
  const int bc = blockIdx.x;                       // b*512 + c
  const float* base = kf + (size_t)bc * 4096;
  const int tid = threadIdx.x;
  float m = -1e30f;
  for (int i = tid; i < 4096; i += 256) m = fmaxf(m, base[i]);
  red[tid] = m; __syncthreads();
  for (int s = 128; s > 0; s >>= 1) {
    if (tid < s) red[tid] = fmaxf(red[tid], red[tid + s]);
    __syncthreads();
  }
  m = red[0]; __syncthreads();
  float sum = 0.f;
  for (int i = tid; i < 4096; i += 256) sum += __expf(base[i] - m);
  red[tid] = sum; __syncthreads();
  for (int s = 128; s > 0; s >>= 1) {
    if (tid < s) red[tid] += red[tid + s];
    __syncthreads();
  }
  const float inv = 1.f / red[0];
  unsigned short* dst = k_bf + (size_t)bc * 4096;
  for (int i = tid; i < 4096; i += 256) dst[i] = f2bf(__expf(base[i] - m) * inv);
}

// ---------------------------------------------------------------------------
// 6a) ctx partials: ctx[bh] = k^T v (64x64, K=4096) with 4-way K split for
//     occupancy (256 workgroups). Fragments load straight from global as
//     b128 chunks (both operands are K-contiguous in memory).
// ---------------------------------------------------------------------------
__global__ __launch_bounds__(512) void k_ctx(const unsigned short* __restrict__ k_bf,
                                             const unsigned short* __restrict__ v_bf,
                                             float* __restrict__ ctx_part) {
  const int bh = blockIdx.x;            // 64
  const int sl = blockIdx.y;            // K slice 0..3
  const int b = bh >> 3, h = bh & 7;
  const int tid  = threadIdx.x;         // 512
  const int lane = tid & 31;
  const int wv   = tid >> 5;            // 0..15
  const int mt = wv >> 2, nt = wv & 3;
  const unsigned short* kp = k_bf + ((size_t)b * 512 + h * 64) * 4096;
  const unsigned short* vp = v_bf + ((size_t)b * 512 + h * 64) * 4096;
  const int row = mt * 16 + (lane & 15);
  const int col = nt * 16 + (lane & 15);
  const int ka  = (lane < 16) ? 0 : 16;
  const int kc  = (lane < 16) ? 0 : 16;
  f32x8 acc = {};
  const int kbeg = sl * 1024, kend = kbeg + 1024;
  for (int k0 = kbeg; k0 < kend; k0 += 32) {
    Frag16 af, bfr;
    const U4* pa = (const U4*)(kp + (size_t)row * 4096 + k0 + ka);
    af.q[0] = pa[0]; af.q[1] = pa[1];          // K and K+8 chunks
    const U4* pb = (const U4*)(vp + (size_t)col * 4096 + k0 + kc);
    bfr.q[0] = pb[0]; bfr.q[1] = pb[1];
    acc = wmma_bf16(af, bfr, acc);
  }
  const int mofs = (lane >> 4) << 3;
  for (int r = 0; r < 8; ++r) {
    int d = mt * 16 + r + mofs;
    int e = nt * 16 + (lane & 15);
    ctx_part[(((size_t)bh * 4 + sl) * 64 + d) * 64 + e] = acc[r];
  }
}

// 6b) reduce 4 K-slices -> bf16 ctx [bh][d][e]
__global__ __launch_bounds__(256) void k_ctx_red(const float* __restrict__ ctx_part,
                                                 unsigned short* __restrict__ ctx_bf) {
  const int idx = blockIdx.x * 256 + threadIdx.x;   // 0 .. 64*64*64-1
  const int bh = idx >> 12;
  const int de = idx & 4095;
  float s = 0.f;
  #pragma unroll
  for (int sl = 0; sl < 4; ++sl)
    s += ctx_part[(((size_t)bh * 4 + sl) << 12) + de];
  ctx_bf[idx] = f2bf(s);
}

// ---------------------------------------------------------------------------
// 7) out = q @ ctx (M = 4096 tokens, K = 64, N = 64), fused SiLU, -> bf16
//    in [b][h*64+d][n] layout. q and ctx are staged pre-transposed in LDS
//    so both WMMA fragments read 16B-contiguous chunks (ds_load_b128).
// ---------------------------------------------------------------------------
__global__ __launch_bounds__(256) void k_attn_out(const unsigned short* __restrict__ q_bf,
                                                  const unsigned short* __restrict__ ctx_bf,
                                                  unsigned short* __restrict__ attn_bf) {
  __shared__ alignas(16) unsigned short Qs[128][72];  // [n][d] (transposed stage)
  __shared__ alignas(16) unsigned short Cs[64][72];   // [e][d] (transposed stage)
  const int bh = blockIdx.y;               // 64
  const int b = bh >> 3, h = bh & 7;
  const int n0  = blockIdx.x * 128;
  const int tid = threadIdx.x;             // 256
  const unsigned short* qb = q_bf + ((size_t)b * 512 + h * 64) * 4096;
  #pragma unroll
  for (int i = 0; i < 32; ++i) {
    int e = tid + i * 256;
    int d = e >> 7, n = e & 127;           // coalesced global read over n
    Qs[n][d] = qb[(size_t)d * 4096 + n0 + n];
  }
  #pragma unroll
  for (int i = 0; i < 16; ++i) {
    int idx = tid + i * 256;
    int d = idx >> 6, e = idx & 63;        // coalesced global read over e
    Cs[e][d] = ctx_bf[(size_t)bh * 4096 + idx];
  }
  __syncthreads();

  const int lane = tid & 31;
  const int wv   = tid >> 5;               // token sub-tile 0..7
  const int row  = wv * 16 + (lane & 15);
  const int ka   = (lane < 16) ? 0 : 16;
  const int kc   = (lane < 16) ? 0 : 16;
  f32x8 acc[4];
  f32x8 zero = {};
  for (int i = 0; i < 4; ++i) acc[i] = zero;

  #pragma unroll
  for (int k0 = 0; k0 < 64; k0 += 32) {
    Frag16 af;
    af.q[0] = *(const U4*)&Qs[row][k0 + ka];
    af.q[1] = *(const U4*)&Qs[row][k0 + 8 + ka];
    #pragma unroll
    for (int nt = 0; nt < 4; ++nt) {
      const int col = nt * 16 + (lane & 15);
      Frag16 bfr;
      const U4* p = (const U4*)&Cs[col][k0 + kc];
      bfr.q[0] = p[0]; bfr.q[1] = p[1];
      acc[nt] = wmma_bf16(af, bfr, acc[nt]);
    }
  }

  const int mofs = (lane >> 4) << 3;
  #pragma unroll
  for (int nt = 0; nt < 4; ++nt)
    #pragma unroll
    for (int r = 0; r < 8; ++r) {
      float v  = acc[nt][r];
      float sv = v * (1.f / (1.f + __expf(-v)));       // SiLU
      int token = n0 + wv * 16 + r + mofs;
      int dch   = nt * 16 + (lane & 15);
      attn_bf[((size_t)b * 512 + h * 64 + dch) * 4096 + token] = f2bf(sv);
    }
}

// ---------------------------------------------------------------------------
// Launch
// ---------------------------------------------------------------------------
extern "C" void kernel_launch(void* const* d_in, const int* in_sizes, int n_in,
                              void* d_out, int out_size, void* d_ws, size_t ws_size,
                              hipStream_t stream) {
  const float* fmap  = (const float*)d_in[0];
  const float* gamma = (const float*)d_in[1];
  const float* wq1   = (const float*)d_in[2];
  const float* wq_dw = (const float*)d_in[3];
  const float* wk1   = (const float*)d_in[4];
  const float* wk_dw = (const float*)d_in[5];
  const float* wv1   = (const float*)d_in[6];
  const float* wv_dw = (const float*)d_in[7];
  const float* w_out = (const float*)d_in[8];

  char* ws = (char*)d_ws;
  const size_t NC = (size_t)8 * 512 * 4096;   // elements per [B][512][4096] tensor
  size_t off = 0;
  auto take = [&](size_t bytes) {
    size_t o = off;
    off = (off + bytes + 255) & ~(size_t)255;
    return o;
  };
  unsigned short* fm_bf   = (unsigned short*)(ws + take(NC * 2));
  float*          qkv1    = (float*)         (ws + take(NC * 3 * 4));
  float*          qf      = (float*)         (ws + take(NC * 4));
  float*          kf      = (float*)         (ws + take(NC * 4));
  unsigned short* v_bf    = (unsigned short*)(ws + take(NC * 2));
  unsigned short* q_bf    = (unsigned short*)(ws + take(NC * 2));
  unsigned short* k_bf    = (unsigned short*)(ws + take(NC * 2));
  float*          ctx_par = (float*)         (ws + take((size_t)64 * 4 * 64 * 64 * 4));
  unsigned short* ctx_bf  = (unsigned short*)(ws + take((size_t)64 * 64 * 64 * 2));
  unsigned short* attn_bf = (unsigned short*)(ws + take(NC * 2));

  k_ln<<<128, 256, 0, stream>>>(fmap, gamma, fm_bf);
  k_gemm_wmma<<<dim3(32, 12, 8), 256, 0, stream>>>(wq1, wk1, wv1, fm_bf, qkv1, 1536);
  k_dwconv<<<dim3(8 * 1536, 16), 256, 0, stream>>>(qkv1, wq_dw, wk_dw, wv_dw, qf, kf, v_bf);
  k_softmax_q<<<dim3(64, 16), 256, 0, stream>>>(qf, q_bf);
  k_softmax_k<<<4096, 256, 0, stream>>>(kf, k_bf);
  k_ctx<<<dim3(64, 4), 512, 0, stream>>>(k_bf, v_bf, ctx_par);
  k_ctx_red<<<1024, 256, 0, stream>>>(ctx_par, ctx_bf);
  k_attn_out<<<dim3(32, 64), 256, 0, stream>>>(q_bf, ctx_bf, attn_bf);
  k_gemm_wmma<<<dim3(32, 4, 8), 256, 0, stream>>>(w_out, w_out, w_out, attn_bf,
                                                  (float*)d_out, 512);
}
